// GDNet_2911987827216
// MI455X (gfx1250) — matrix operand order repaired and checked by reading, not compiled, not run on hardware
//
#include <hip/hip_runtime.h>
#include <math.h>

// ---------------------------------------------------------------------------
// GDNet (GAT + dense_diff_pool) forward for gfx1250.
// All matmuls run through v_wmma_f32_16x16x32_f16 (f32 accumulate).
// Attention is never materialized: row-stats pass + fused recompute-apply.
// All in-loop bounds handling is branchless (clamped addresses + selects);
// full tiles (all of layer 0) take an unguarded vectorized path.
// ---------------------------------------------------------------------------

typedef __attribute__((ext_vector_type(16))) _Float16 v16h;
typedef __attribute__((ext_vector_type(8)))  float    v8f;

#define ALPHA_   0.2f
#define EPS_     1e-15f
#define NEG_INF_ -9e15f

// ISA 7.12.2 16-bit A-matrix 16x32 layout: lane m=l&15,
// k = (lane>=16?8:0) + {j for j<8 ; 16+(j-8) for j>=8}
__device__ __forceinline__ int kidxA(int lane, int j) {
  return ((lane >= 16) ? 8 : 0) + ((j < 8) ? j : (8 + j));
}
// 16-bit B-matrix 32x16: lane n=l&15; k = (lane>=16?8:0) + {j for j<8 ; 16+(j-8) for j>=8}
__device__ __forceinline__ int kidxB(int lane, int j) {
  return ((lane >= 16) ? 8 : 0) + ((j < 8) ? j : (8 + j));
}
__device__ __forceinline__ float lrelu(float x) { return x >= 0.f ? x : ALPHA_ * x; }

// ---------------------------------------------------------------------------
// Batched GEMM: C[b] = (relu?) opA(A[b]) x B[b]. One wave per 16x16 tile.
// ---------------------------------------------------------------------------
template <int TRANSA>
__global__ void gemm_wmma(const float* __restrict__ Abase,
                          const float* __restrict__ Bbase,
                          float* __restrict__ Cbase,
                          int M, int N, int K,
                          long long sA, long long sB, long long sC,
                          int lda, int ldb, int ldc, int relu) {
  const int lane = threadIdx.x;
  const int tn = blockIdx.x, tm = blockIdx.y, b = blockIdx.z;
  const float* A  = Abase + (long long)b * sA;
  const float* Bm = Bbase + (long long)b * sB;
  float*       C  = Cbase + (long long)b * sC;
  const int mrow  = tm * 16 + (lane & 15);
  const int ncol  = tn * 16 + (lane & 15);
  const int baseA = (lane >= 16) ? 8 : 0;
  const int baseB = (lane >= 16) ? 8 : 0;
  const bool fast = (tm * 16 + 16 <= M) && (tn * 16 + 16 <= N) && ((K & 31) == 0);
  v8f acc = {};
  if (fast) {
    for (int k0 = 0; k0 < K; k0 += 32) {
      v16h av, bv;
      if (TRANSA) {
#pragma unroll
        for (int j = 0; j < 16; ++j)
          av[j] = (_Float16)A[(long long)(k0 + kidxA(lane, j)) * lda + mrow];
      } else {
        const float* ar = A + (long long)mrow * lda + k0 + baseA;
#pragma unroll
        for (int j = 0; j < 8; ++j) av[j]     = (_Float16)ar[j];
#pragma unroll
        for (int j = 0; j < 8; ++j) av[8 + j] = (_Float16)ar[16 + j];
      }
#pragma unroll
      for (int j = 0; j < 16; ++j)
        bv[j] = (_Float16)Bm[(long long)(k0 + baseB + ((j < 8) ? j : (8 + j))) * ldb + ncol];
      acc = __builtin_amdgcn_wmma_f32_16x16x32_f16(false, av, false, bv,
                                                   (short)0, acc, false, false);
    }
    const int rbase = tm * 16 + ((lane >= 16) ? 8 : 0);
#pragma unroll
    for (int v = 0; v < 8; ++v) {
      float val = acc[v];
      if (relu) val = fmaxf(val, 0.f);
      C[(long long)(rbase + v) * ldc + ncol] = val;
    }
  } else {
    const bool mok = mrow < M;
    const bool nok = ncol < N;
    const int  mc  = mok ? mrow : M - 1;
    const int  nc  = nok ? ncol : N - 1;
    for (int k0 = 0; k0 < K; k0 += 32) {
      v16h av, bv;
#pragma unroll
      for (int j = 0; j < 16; ++j) {
        int kk = k0 + kidxA(lane, j);
        int kc = kk < K ? kk : K - 1;
        float v = TRANSA ? A[(long long)kc * lda + mc] : A[(long long)mc * lda + kc];
        av[j] = (_Float16)((mok && kk < K) ? v : 0.f);
      }
#pragma unroll
      for (int j = 0; j < 16; ++j) {
        int kk = k0 + kidxB(lane, j);
        int kc = kk < K ? kk : K - 1;
        float v = Bm[(long long)kc * ldb + nc];
        bv[j] = (_Float16)((nok && kk < K) ? v : 0.f);
      }
      acc = __builtin_amdgcn_wmma_f32_16x16x32_f16(false, av, false, bv,
                                                   (short)0, acc, false, false);
    }
    const int rbase = tm * 16 + ((lane >= 16) ? 8 : 0);
#pragma unroll
    for (int v = 0; v < 8; ++v) {
      int r = rbase + v;
      if (r < M && nok) {
        float val = acc[v];
        if (relu) val = fmaxf(val, 0.f);
        C[(long long)r * ldc + ncol] = val;
      }
    }
  }
}

// Convert + transpose: out[b][f][n] (f16) = in[b][n][f] (f32)
__global__ void cvt_transpose_f16(const float* __restrict__ in, _Float16* __restrict__ out,
                                  int Nn, int Fo) {
  long long b = blockIdx.z;
  int n = blockIdx.x * 32 + threadIdx.x;
  int f = blockIdx.y * 8 + threadIdx.y;
  if (n < Nn && f < Fo)
    out[(b * Fo + f) * Nn + n] = (_Float16)in[(b * Nn + n) * Fo + f];
}

// f1[r] = Wh[r,:].a[:Fo], f2[r] = Wh[r,:].a[Fo:]  (warp per row)
__global__ void fvec_kernel(const float* __restrict__ Wh, const float* __restrict__ a,
                            float* __restrict__ f1, float* __restrict__ f2,
                            int rows, int Fo) {
  int warp = (int)((blockIdx.x * blockDim.x + threadIdx.x) >> 5);
  int lane = threadIdx.x & 31;
  if (warp >= rows) return;
  const float* w = Wh + (long long)warp * Fo;
  float s1 = 0.f, s2 = 0.f;
  for (int f = lane; f < Fo; f += 32) { float x = w[f]; s1 += x * a[f]; s2 += x * a[Fo + f]; }
  for (int o = 16; o > 0; o >>= 1) { s1 += __shfl_down(s1, o, 32); s2 += __shfl_down(s2, o, 32); }
  if (lane == 0) { f1[warp] = s1; f2[warp] = s2; }
}

// Per attention row: masked max and sum(exp(e - max)).
__global__ void rowstats_kernel(const float* __restrict__ A,
                                const float* __restrict__ f1,
                                const float* __restrict__ f2,
                                float* __restrict__ rmax, float* __restrict__ rsum,
                                int Nn) {
  __shared__ float red[256];
  int i = blockIdx.x, b = blockIdx.y, t = threadIdx.x;
  const float* arow = A + ((long long)b * Nn + i) * Nn;
  const float* f2b  = f2 + (long long)b * Nn;
  float fi = f1[(long long)b * Nn + i];
  float m = NEG_INF_;
  for (int j = t; j < Nn; j += blockDim.x) {
    float e = lrelu(fi + f2b[j]);
    m = fmaxf(m, (arow[j] > 0.f) ? e : NEG_INF_);
  }
  red[t] = m; __syncthreads();
  for (int s = 128; s > 0; s >>= 1) { if (t < s) red[t] = fmaxf(red[t], red[t + s]); __syncthreads(); }
  m = red[0]; __syncthreads();
  float s = 0.f;
  for (int j = t; j < Nn; j += blockDim.x) {
    float e = __expf(lrelu(fi + f2b[j]) - m);
    s += (arow[j] > 0.f) ? e : 0.f;
  }
  red[t] = s; __syncthreads();
  for (int st = 128; st > 0; st >>= 1) { if (t < st) red[t] += red[t + st]; __syncthreads(); }
  if (t == 0) { rmax[(long long)b * Nn + i] = m; rsum[(long long)b * Nn + i] = red[0]; }
}

// O = (relu?) softmax-attention @ Wh.  Attention fragment rebuilt in registers
// per K-step; WhT is pre-converted f16 [Fo,Nn] so B-fragments are contiguous.
__global__ void attn_apply_wmma(const float* __restrict__ A, const _Float16* __restrict__ WhT,
                                const float* __restrict__ f1, const float* __restrict__ f2,
                                const float* __restrict__ rmax, const float* __restrict__ rsum,
                                float* __restrict__ O, int Nn, int Fo, int relu) {
  const int lane = threadIdx.x;
  const int ti = blockIdx.y, b = blockIdx.z;
  const int tf0 = blockIdx.x * 4;
  const float*    Ab   = A   + (long long)b * Nn * Nn;
  const _Float16* WhTb = WhT + (long long)b * Fo * Nn;
  const float*    f2b  = f2  + (long long)b * Nn;
  const int m = ti * 16 + (lane & 15);
  const int baseA = (lane >= 16) ? 8 : 0;
  const int baseB = (lane >= 16) ? 8 : 0;
  const bool mok = m < Nn;
  const int  mc  = mok ? m : Nn - 1;
  const float fi = f1[(long long)b * Nn + mc];
  const float mi = rmax[(long long)b * Nn + mc];
  const float sm = rsum[(long long)b * Nn + mc];
  const float inv = (sm > 0.f) ? 1.f / sm : 0.f;
  const bool fast = ((Nn & 31) == 0) && (ti * 16 + 16 <= Nn) && (blockIdx.x * 64 + 64 <= Fo);
  v8f acc[4] = {{}, {}, {}, {}};
  if (fast) {
    for (int j0 = 0; j0 < Nn; j0 += 32) {
      const float* ar = Ab  + (long long)m * Nn + j0 + baseA;
      const float* fr = f2b + j0 + baseA;
      v16h av;
#pragma unroll
      for (int j = 0; j < 8; ++j) {
        float a = ar[j];
        float p = __expf(lrelu(fi + fr[j]) - mi) * inv;
        av[j] = (_Float16)((a > 0.f) ? p : 0.f);
      }
#pragma unroll
      for (int j = 0; j < 8; ++j) {
        float a = ar[16 + j];
        float p = __expf(lrelu(fi + fr[16 + j]) - mi) * inv;
        av[8 + j] = (_Float16)((a > 0.f) ? p : 0.f);
      }
#pragma unroll
      for (int ft = 0; ft < 4; ++ft) {
        const int col = (tf0 + ft) * 16 + (lane & 15);
        const _Float16* bp = WhTb + (long long)col * Nn + j0 + baseB;
        v16h bv;
#pragma unroll
        for (int j = 0; j < 16; ++j) bv[j] = bp[(j < 8) ? j : (8 + j)];
        acc[ft] = __builtin_amdgcn_wmma_f32_16x16x32_f16(false, av, false, bv,
                                                         (short)0, acc[ft], false, false);
      }
    }
    const int rbase = ti * 16 + ((lane >= 16) ? 8 : 0);
#pragma unroll
    for (int ft = 0; ft < 4; ++ft) {
      const int col = (tf0 + ft) * 16 + (lane & 15);
#pragma unroll
      for (int v = 0; v < 8; ++v) {
        float val = acc[ft][v];
        if (relu) val = fmaxf(val, 0.f);
        O[((long long)b * Nn + rbase + v) * Fo + col] = val;
      }
    }
  } else {
    for (int j0 = 0; j0 < Nn; j0 += 32) {
      v16h av;
#pragma unroll
      for (int j = 0; j < 16; ++j) {
        int kk = j0 + kidxA(lane, j);
        int kc = kk < Nn ? kk : Nn - 1;
        float a = Ab[(long long)mc * Nn + kc];
        float p = __expf(lrelu(fi + f2b[kc]) - mi) * inv;
        av[j] = (_Float16)((mok && kk < Nn && a > 0.f) ? p : 0.f);
      }
#pragma unroll
      for (int ft = 0; ft < 4; ++ft) {
        const int col = (tf0 + ft) * 16 + (lane & 15);
        const int cc  = col < Fo ? col : Fo - 1;
        v16h bv;
#pragma unroll
        for (int j = 0; j < 16; ++j) {
          int kk = j0 + kidxB(lane, j);
          int kc = kk < Nn ? kk : Nn - 1;
          _Float16 v = WhTb[(long long)cc * Nn + kc];
          bv[j] = (col < Fo && kk < Nn) ? v : (_Float16)0.f;
        }
        acc[ft] = __builtin_amdgcn_wmma_f32_16x16x32_f16(false, av, false, bv,
                                                         (short)0, acc[ft], false, false);
      }
    }
    const int rbase = ti * 16 + ((lane >= 16) ? 8 : 0);
#pragma unroll
    for (int ft = 0; ft < 4; ++ft) {
      const int col = (tf0 + ft) * 16 + (lane & 15);
#pragma unroll
      for (int v = 0; v < 8; ++v) {
        int r = rbase + v;
        if (r < Nn && col < Fo) {
          float val = acc[ft][v];
          if (relu) val = fmaxf(val, 0.f);
          O[((long long)b * Nn + r) * Fo + col] = val;
        }
      }
    }
  }
}

// S = softmax(relu(Sraw)) over c; accumulate entropy sum.
__global__ void smax_ent_kernel(const float* __restrict__ Sraw, float* __restrict__ S,
                                float* __restrict__ ent_acc, int rows, int c) {
  int r = blockIdx.x * blockDim.x + threadIdx.x;
  if (r >= rows) return;
  const float* in = Sraw + (long long)r * c;
  float*      out = S    + (long long)r * c;
  float m = 0.f;
  for (int k = 0; k < c; ++k) m = fmaxf(m, in[k]);
  float sum = 0.f;
  for (int k = 0; k < c; ++k) sum += __expf(fmaxf(in[k], 0.f) - m);
  float invs = 1.f / sum;
  float ent = 0.f;
  for (int k = 0; k < c; ++k) {
    float p = __expf(fmaxf(in[k], 0.f) - m) * invs;
    out[k] = p;
    ent += -p * __logf(p + EPS_);
  }
  atomicAdd(ent_acc, ent);
}

// sum((A - S S^T)^2): WMMA tile of S@S^T, squared diff vs A, shfl+atomic reduce.
__global__ void link_wmma(const float* __restrict__ A, const float* __restrict__ S,
                          float* __restrict__ sumsq_acc, int Nn, int c) {
  const int lane = threadIdx.x;
  const int tn = blockIdx.x, tm = blockIdx.y, b = blockIdx.z;
  const float* Ab = A + (long long)b * Nn * Nn;
  const float* Sb = S + (long long)b * Nn * c;
  const int m = tm * 16 + (lane & 15);
  const int n = tn * 16 + (lane & 15);
  const int baseA = (lane >= 16) ? 8 : 0;
  const int baseB = (lane >= 16) ? 8 : 0;
  const bool fast = ((c & 31) == 0) && ((Nn & 15) == 0);
  v8f acc = {};
  if (fast) {
    for (int k0 = 0; k0 < c; k0 += 32) {
      const float* ar = Sb + (long long)m * c + k0 + baseA;
      const float* br = Sb + (long long)n * c + k0 + baseB;
      v16h av, bv;
#pragma unroll
      for (int j = 0; j < 8; ++j) { av[j] = (_Float16)ar[j]; av[8 + j] = (_Float16)ar[16 + j]; }
#pragma unroll
      for (int j = 0; j < 16; ++j) bv[j] = (_Float16)br[(j < 8) ? j : (8 + j)];
      acc = __builtin_amdgcn_wmma_f32_16x16x32_f16(false, av, false, bv,
                                                   (short)0, acc, false, false);
    }
  } else {
    const bool mok = m < Nn, nok = n < Nn;
    const int  mc = mok ? m : Nn - 1, nc2 = nok ? n : Nn - 1;
    for (int k0 = 0; k0 < c; k0 += 32) {
      v16h av, bv;
#pragma unroll
      for (int j = 0; j < 16; ++j) {
        int kk = k0 + kidxA(lane, j);
        int kc = kk < c ? kk : c - 1;
        float v = Sb[(long long)mc * c + kc];
        av[j] = (_Float16)((mok && kk < c) ? v : 0.f);
      }
#pragma unroll
      for (int j = 0; j < 16; ++j) {
        int kk = k0 + kidxB(lane, j);
        int kc = kk < c ? kk : c - 1;
        float v = Sb[(long long)nc2 * c + kc];
        bv[j] = (_Float16)((nok && kk < c) ? v : 0.f);
      }
      acc = __builtin_amdgcn_wmma_f32_16x16x32_f16(false, av, false, bv,
                                                   (short)0, acc, false, false);
    }
  }
  const int rbase = tm * 16 + ((lane >= 16) ? 8 : 0);
  float local = 0.f;
#pragma unroll
  for (int v = 0; v < 8; ++v) {
    int r = rbase + v;
    if (r < Nn && n < Nn) {
      float d = Ab[(long long)r * Nn + n] - acc[v];
      local += d * d;
    }
  }
  for (int off = 16; off > 0; off >>= 1) local += __shfl_down(local, off, 32);
  if (lane == 0) atomicAdd(sumsq_acc, local);
}

__global__ void init_acc_kernel(float* acc) { if (threadIdx.x < 8) acc[threadIdx.x] = 0.f; }

// logits (16x10), A3 copy, link/ent totals.
__global__ void final_kernel(const float* __restrict__ X3, const float* __restrict__ A3,
                             const float* __restrict__ Wd, const float* __restrict__ bd,
                             const float* __restrict__ acc, float* __restrict__ out) {
  int t = blockIdx.x * blockDim.x + threadIdx.x;
  if (t < 160) {
    int b = t / 10, cls = t % 10;
    float s = bd[cls];
    const float* x = X3 + b * 256;
    for (int f = 0; f < 256; ++f) s += x[f] * Wd[f * 10 + cls];
    out[t] = s;
  } else if (t < 176) {
    out[t] = A3[t - 160];
  } else if (t == 176) {
    out[176] = sqrtf(acc[0]) / (16.f * 1024.f * 1024.f)
             + sqrtf(acc[1]) / (16.f * 64.f * 64.f)
             + sqrtf(acc[2]) / (16.f * 8.f * 8.f);
  } else if (t == 177) {
    out[177] = acc[3] / (16.f * 1024.f) + acc[4] / (16.f * 64.f) + acc[5] / (16.f * 8.f);
  }
}

// ---------------------------------------------------------------------------
extern "C" void kernel_launch(void* const* d_in, const int* in_sizes, int n_in,
                              void* d_out, int out_size, void* d_ws, size_t ws_size,
                              hipStream_t stream) {
  (void)in_sizes; (void)n_in; (void)out_size; (void)ws_size;
  const float* X   = (const float*)d_in[0];
  const float* A0  = (const float*)d_in[1];
  const float* Wd  = (const float*)d_in[14];
  const float* bd  = (const float*)d_in[15];
  const int B = 16;

  float* wsp = (float*)d_ws;
  size_t off = 0;
  float* WHG = wsp + off; off += (size_t)B * 1024 * 256;
  float* WHS = wsp + off; off += (size_t)B * 1024 * 64;
  float* Zb  = wsp + off; off += (size_t)B * 1024 * 256;
  float* Sb  = wsp + off; off += (size_t)B * 1024 * 64;
  float* Tb  = wsp + off; off += (size_t)B * 1024 * 64;
  float* f1g = wsp + off; off += (size_t)B * 1024;
  float* f2g = wsp + off; off += (size_t)B * 1024;
  float* mg  = wsp + off; off += (size_t)B * 1024;
  float* sg  = wsp + off; off += (size_t)B * 1024;
  float* f1s = wsp + off; off += (size_t)B * 1024;
  float* f2s = wsp + off; off += (size_t)B * 1024;
  float* ms  = wsp + off; off += (size_t)B * 1024;
  float* ss2 = wsp + off; off += (size_t)B * 1024;
  float* X1  = wsp + off; off += (size_t)B * 64 * 256;
  float* A1  = wsp + off; off += (size_t)B * 64 * 64;
  float* X2  = wsp + off; off += (size_t)B * 8 * 256;
  float* A2  = wsp + off; off += (size_t)B * 8 * 8;
  float* X3  = wsp + off; off += (size_t)B * 256;
  float* A3  = wsp + off; off += (size_t)B;
  float* ACC = wsp + off; off += 8;   // [0..2]=link sumsq, [3..5]=ent sums
  _Float16* WHG16T = (_Float16*)(wsp + off); off += (size_t)B * 256 * 1024 / 2; // f16 [B,256,Nn]
  _Float16* WHS16T = (_Float16*)(wsp + off); off += (size_t)B * 64  * 1024 / 2; // f16 [B,c,Nn]

  init_acc_kernel<<<1, 32, 0, stream>>>(ACC);

  const int dims[3][3] = {{1024, 128, 64}, {64, 256, 8}, {8, 256, 1}};
  const float* Xcur = X;
  const float* Acur = A0;
  float* Xn[3] = {X1, X2, X3};
  float* An[3] = {A1, A2, A3};

  for (int l = 0; l < 3; ++l) {
    const int Nn  = dims[l][0];
    const int Fin = dims[l][1];
    const int c   = dims[l][2];
    const float* Wg  = (const float*)d_in[2 + 4 * l];
    const float* ag  = (const float*)d_in[3 + 4 * l];
    const float* Wsm = (const float*)d_in[4 + 4 * l];
    const float* asv = (const float*)d_in[5 + 4 * l];
    const int tM = (Nn + 15) / 16;
    const int tC = (c + 15) / 16;
    const int rows = B * Nn;

    // Whg = Xcur @ Wg  [B,Nn,256]
    gemm_wmma<0><<<dim3(16, tM, B), 32, 0, stream>>>(
        Xcur, Wg, WHG, Nn, 256, Fin,
        (long long)Nn * Fin, 0, (long long)Nn * 256, Fin, 256, 256, 0);
    // Whs = Xcur @ Ws  [B,Nn,c]
    gemm_wmma<0><<<dim3(tC, tM, B), 32, 0, stream>>>(
        Xcur, Wsm, WHS, Nn, c, Fin,
        (long long)Nn * Fin, 0, (long long)Nn * c, Fin, c, c, 0);
    // f16 transposed copies for attention B-fragments
    cvt_transpose_f16<<<dim3((Nn + 31) / 32, (256 + 7) / 8, B), dim3(32, 8), 0, stream>>>(
        WHG, WHG16T, Nn, 256);
    cvt_transpose_f16<<<dim3((Nn + 31) / 32, (c + 7) / 8, B), dim3(32, 8), 0, stream>>>(
        WHS, WHS16T, Nn, c);
    // attention coefficients
    fvec_kernel<<<(rows * 32 + 255) / 256, 256, 0, stream>>>(WHG, ag, f1g, f2g, rows, 256);
    fvec_kernel<<<(rows * 32 + 255) / 256, 256, 0, stream>>>(WHS, asv, f1s, f2s, rows, c);
    rowstats_kernel<<<dim3(Nn, B), 256, 0, stream>>>(Acur, f1g, f2g, mg, sg, Nn);
    rowstats_kernel<<<dim3(Nn, B), 256, 0, stream>>>(Acur, f1s, f2s, ms, ss2, Nn);
    // Z = relu(att_g @ Whg)
    attn_apply_wmma<<<dim3(4, tM, B), 32, 0, stream>>>(
        Acur, WHG16T, f1g, f2g, mg, sg, Zb, Nn, 256, 1);
    // Sraw = relu(att_s @ Whs) -> Tb
    attn_apply_wmma<<<dim3((c + 63) / 64, tM, B), 32, 0, stream>>>(
        Acur, WHS16T, f1s, f2s, ms, ss2, Tb, Nn, c, 1);
    // S = softmax(Sraw) over c; entropy sum
    smax_ent_kernel<<<(rows + 255) / 256, 256, 0, stream>>>(Tb, Sb, ACC + 3 + l, rows, c);
    // link loss: sum((A - S S^T)^2)
    link_wmma<<<dim3(tM, tM, B), 32, 0, stream>>>(Acur, Sb, ACC + l, Nn, c);
    // Xnext = S^T @ Z  [B,c,256]
    gemm_wmma<1><<<dim3(16, tC, B), 32, 0, stream>>>(
        Sb, Zb, Xn[l], c, 256, Nn,
        (long long)Nn * c, (long long)Nn * 256, (long long)c * 256, c, 256, 256, 1);
    // T = A @ S  [B,Nn,c]
    gemm_wmma<0><<<dim3(tC, tM, B), 32, 0, stream>>>(
        Acur, Sb, Tb, Nn, c, Nn,
        (long long)Nn * Nn, (long long)Nn * c, (long long)Nn * c, Nn, c, c, 0);
    // Anext = S^T @ T  [B,c,c]
    gemm_wmma<1><<<dim3(tC, tC, B), 32, 0, stream>>>(
        Sb, Tb, An[l], c, c, Nn,
        (long long)Nn * c, (long long)Nn * c, (long long)c * c, c, c, c, 1);

    Xcur = Xn[l];
    Acur = An[l];
  }

  final_kernel<<<1, 192, 0, stream>>>(X3, A3, Wd, bd, ACC, (float*)d_out);
}
